// SimpleTransformerBlock_49031346651641
// MI455X (gfx1250) — matrix operand order repaired and checked by reading, not compiled, not run on hardware
//
#include <hip/hip_runtime.h>
#include <hip/hip_bf16.h>
#include <cstdint>

// ---------------------------------------------------------------------------
// CDNA5 / gfx1250 transformer block: bf16 WMMA GEMMs + flash attention,
// with async-to-LDS copies (ASYNCcnt) and a Tensor Data Mover load (TENSORcnt).
// ---------------------------------------------------------------------------

typedef __attribute__((ext_vector_type(16))) __bf16 v16bf;
typedef __attribute__((ext_vector_type(8)))  __bf16 v8bf;
typedef __attribute__((ext_vector_type(8)))  float  v8f;
typedef __attribute__((ext_vector_type(4)))  unsigned v4u;
typedef __attribute__((ext_vector_type(8)))  unsigned v8u;

union ABFrag { v16bf v; v8bf h[2]; };

__device__ __forceinline__ unsigned short f2bf(float f) {
  unsigned u = __float_as_uint(f);
  u += 0x7fffu + ((u >> 16) & 1u);        // round-to-nearest-even
  return (unsigned short)(u >> 16);
}

__device__ __forceinline__ v8f wmma_bf16(const ABFrag& a, const ABFrag& b, v8f c) {
  // D = A(16x32 bf16) * B(32x16 bf16) + C(16x16 f32)
  return __builtin_amdgcn_wmma_f32_16x16x32_bf16(false, a.v, false, b.v,
                                                 (short)0, c, false, false);
}

// Async global->LDS 16B copy (ASYNCcnt tracked, no VGPR round trip).
// the vdst VGPR holds the LDS byte address; generic-pointer low 32 bits give the
// workgroup-relative LDS offset.
__device__ __forceinline__ void async_b128(const unsigned short* lds_dst,
                                           const unsigned short* gsrc) {
  unsigned off = (unsigned)(unsigned long long)(const void*)lds_dst;
  asm volatile("global_load_async_to_lds_b128 %0, %1, off"
               :: "v"(off), "v"(gsrc) : "memory");
}
__device__ __forceinline__ void wait_async0() {
  asm volatile("s_wait_asynccnt 0" ::: "memory");
}

// ---------------------------------------------------------------------------
// Tiled GEMM: C[M,N] = A[M,K] @ W[K,N] (+bias, + epilogue)
// Block tile 128x128, K-step 32, 8 waves (256 thr): wave = 32(M) x 64(N).
// A staged row-major in LDS (stride 40 halves), W staged transposed
// (Bs[n][k], stride 40) so both fragment loads are contiguous 16B b128 reads.
// When A is already bf16 (FFN2), the 128x32 A tile is fetched by the TDM:
// 2-D descriptor, data_size=2B, tile 32x128, LDS padding 4 DW per 16 DW
// -> row stride 40 halves, identical to the manual layout.
// ---------------------------------------------------------------------------
enum { EPI_QKV = 0, EPI_GELU = 1, EPI_RES = 2 };

template <int EPI, bool ABF16>
__launch_bounds__(256)
__global__ void gemm_wmma(const void* __restrict__ Aptr,
                          const float* __restrict__ W,
                          const float* __restrict__ bias,
                          const float* __restrict__ resid,
                          void* __restrict__ out0,
                          void* __restrict__ out1,
                          void* __restrict__ out2,
                          int M, int N, int K) {
  __shared__ unsigned short As[128 * 40];
  __shared__ unsigned short Bs[128 * 40];

  const int tid  = threadIdx.x;
  const int wave = tid >> 5, lane = tid & 31;
  const int half = lane >> 4, l16 = lane & 15;
  const int bm = blockIdx.x * 128, bn = blockIdx.y * 128;
  const int wm = (wave >> 1) * 32, wn = (wave & 1) * 64;

  v8f acc[2][4];
#pragma unroll
  for (int i = 0; i < 2; i++)
#pragma unroll
    for (int j = 0; j < 4; j++)
#pragma unroll
      for (int r = 0; r < 8; r++) acc[i][j][r] = 0.f;

  const int arow = tid >> 1, acolb = (tid & 1) * 16;   // A stage: 16 elems/thr
  const int wkk = tid >> 3, wnb = (tid & 7) * 16;      // W stage: 16 elems/thr

  for (int k0 = 0; k0 < K; k0 += 32) {
    // ---- stage A tile (128x32) as bf16 ----
    if (!ABF16) {
      const float* Ag = (const float*)Aptr + (size_t)(bm + arow) * K + k0 + acolb;
      if (k0 + 32 < K) __builtin_prefetch(Ag + 32, 0, 1);   // global_prefetch
#pragma unroll
      for (int i = 0; i < 4; i++) {
        float4 f = *(const float4*)(Ag + 4 * i);
        uint2 pk;
        pk.x = (unsigned)f2bf(f.x) | ((unsigned)f2bf(f.y) << 16);
        pk.y = (unsigned)f2bf(f.z) | ((unsigned)f2bf(f.w) << 16);
        *(uint2*)&As[arow * 40 + acolb + 4 * i] = pk;
      }
    } else if (wave == 0) {
      // Tensor Data Mover: h[bm.., k0..] 128x32 bf16 tile -> As (stride 40)
      unsigned long long ga = (unsigned long long)(const void*)
          ((const unsigned short*)Aptr + (size_t)bm * K + k0);
      v4u g0;
      g0.x = 1u;                                    // count=1, user mode
      g0.y = (unsigned)(unsigned long long)(void*)&As[0];  // lds_addr
      g0.z = (unsigned)ga;                          // global_addr[31:0]
      g0.w = (unsigned)(ga >> 32) | (2u << 30);     // global_addr[56:32], type=2
      const unsigned td0 = (unsigned)K, td1 = (unsigned)M, st0 = (unsigned)K;
      v8u g1;
      g1.s0 = (1u << 16)      // data_size = 2 bytes
            | (1u << 20)      // pad_enable
            | (3u << 22)      // pad_interval: 16 DWORDs
            | (3u << 25);     // pad_amount:   4 DWORDs  -> stride 40 halves
      g1.s1 = (td0 & 0xffffu) << 16;                  // tensor_dim0 lo
      g1.s2 = (td0 >> 16) | ((td1 & 0xffffu) << 16);  // tensor_dim0 hi, dim1 lo
      g1.s3 = (td1 >> 16) | (32u << 16);              // dim1 hi, tile_dim0=32
      g1.s4 = 128u;                                   // tile_dim1=128, tile_dim2=0
      g1.s5 = st0;                                    // tensor_dim0_stride lo
      g1.s6 = 0u;
      g1.s7 = 0u;
      asm volatile("tensor_load_to_lds %0, %1" :: "s"(g0), "s"(g1) : "memory");
    }
    // ---- stage W tile (32x128) transposed: Bs[n][k] ----
    {
      const float* Wg = W + (size_t)(k0 + wkk) * N + bn + wnb;
      if (k0 + 32 < K) __builtin_prefetch(Wg + (size_t)32 * N, 0, 1);
#pragma unroll
      for (int i = 0; i < 16; i += 4) {
        float4 f = *(const float4*)(Wg + i);
        Bs[(wnb + i + 0) * 40 + wkk] = f2bf(f.x);
        Bs[(wnb + i + 1) * 40 + wkk] = f2bf(f.y);
        Bs[(wnb + i + 2) * 40 + wkk] = f2bf(f.z);
        Bs[(wnb + i + 3) * 40 + wkk] = f2bf(f.w);
      }
    }
    if (ABF16 && wave == 0) __builtin_amdgcn_s_wait_tensorcnt(0);
    __syncthreads();

    // ---- fragments & 8 WMMAs ----
    ABFrag a[2], b[4];
#pragma unroll
    for (int tm = 0; tm < 2; tm++) {
      const unsigned short* p = &As[(wm + tm * 16 + l16) * 40];
      a[tm].h[0] = *(const v8bf*)(p + half * 8);        // K = half*8 .. +7
      a[tm].h[1] = *(const v8bf*)(p + 16 + half * 8);   // K = 16+half*8 .. +7
    }
#pragma unroll
    for (int tn = 0; tn < 4; tn++) {
      const unsigned short* p = &Bs[(wn + tn * 16 + l16) * 40];
      b[tn].h[0] = *(const v8bf*)(p + half * 8);        // K = half*8 .. +7
      b[tn].h[1] = *(const v8bf*)(p + 16 + half * 8);   // K = 16+half*8 .. +7
    }
#pragma unroll
    for (int tm = 0; tm < 2; tm++)
#pragma unroll
      for (int tn = 0; tn < 4; tn++)
        acc[tm][tn] = wmma_bf16(a[tm], b[tn], acc[tm][tn]);
    __syncthreads();
  }

  // ---- epilogue ----
#pragma unroll
  for (int tm = 0; tm < 2; tm++) {
#pragma unroll
    for (int tn = 0; tn < 4; tn++) {
      const int nglob = bn + wn + tn * 16 + l16;
      const float bv = bias ? bias[nglob] : 0.f;
      if (EPI == EPI_QKV) {
        const int mat = nglob >> 10, nn = nglob & 1023;
        const int head = nn >> 6, dh = nn & 63;
        unsigned short* dst =
            (unsigned short*)(mat == 0 ? out0 : (mat == 1 ? out1 : out2));
        const float sc = (mat == 0) ? 0.125f : 1.0f;  // fold 1/sqrt(Dh) into Q
#pragma unroll
        for (int r = 0; r < 8; r++) {
          const int mglob = bm + wm + tm * 16 + r + 8 * half;
          const int bb = mglob >> 12, s = mglob & 4095;
          dst[(((size_t)bb * 16 + head) * 4096 + s) * 64 + dh] =
              f2bf((acc[tm][tn][r] + bv) * sc);
        }
      } else if (EPI == EPI_GELU) {
        unsigned short* dst = (unsigned short*)out0;
#pragma unroll
        for (int r = 0; r < 8; r++) {
          const int mglob = bm + wm + tm * 16 + r + 8 * half;
          float v = acc[tm][tn][r] + bv;
          float g = 0.5f * v * (1.0f + erff(v * 0.70710678118654752f));
          dst[(size_t)mglob * N + nglob] = f2bf(g);
        }
      } else {  // EPI_RES
        float* dst = (float*)out0;
#pragma unroll
        for (int r = 0; r < 8; r++) {
          const int mglob = bm + wm + tm * 16 + r + 8 * half;
          dst[(size_t)mglob * N + nglob] =
              acc[tm][tn][r] + bv + resid[(size_t)mglob * N + nglob];
        }
      }
    }
  }
}

// ---------------------------------------------------------------------------
// Flash attention, Dh=64. One WG = 128 threads (4 waves) handles 64 queries
// of one (b,h). Q pre-scaled by 1/8. K-blocks of 32 staged in LDS via
// GLOBAL_LOAD_ASYNC_TO_LDS_B128; V staged transposed. P goes C-layout -> LDS
// -> A-layout for the PV WMMAs.
// ---------------------------------------------------------------------------
__launch_bounds__(128)
__global__ void flash_attn(const unsigned short* __restrict__ Qb,
                           const unsigned short* __restrict__ Kb,
                           const unsigned short* __restrict__ Vb,
                           float* __restrict__ O, int H, int S) {
  __shared__ unsigned short Klds[32 * 72];        // [key][dh], stride 72
  __shared__ unsigned short Vt[64 * 40];          // [dh][key], stride 40
  __shared__ unsigned short Plds[4 * 16 * 40];    // per-wave 16x32 P tile

  const int tid  = threadIdx.x;
  const int wave = tid >> 5, lane = tid & 31;
  const int half = lane >> 4, l16 = lane & 15;
  const int nq = S / 64;
  const int qblk = blockIdx.x % nq;
  const int bh = blockIdx.x / nq;                 // b*H + h
  const int bb = bh / H, hh = bh % H;

  const unsigned short* Qh = Qb + (size_t)bh * S * 64;
  const unsigned short* Kh = Kb + (size_t)bh * S * 64;
  const unsigned short* Vh = Vb + (size_t)bh * S * 64;

  // Q fragments for this wave's 16 rows, both 32-wide K-steps of Dh=64.
  const int qrow = qblk * 64 + wave * 16 + l16;
  ABFrag qa[2];
#pragma unroll
  for (int ks = 0; ks < 2; ks++) {
    const unsigned short* qp = Qh + (size_t)qrow * 64 + ks * 32 + half * 8;
    qa[ks].h[0] = *(const v8bf*)qp;
    qa[ks].h[1] = *(const v8bf*)(qp + 16);
  }

  float mstat[8], lstat[8];
  v8f oacc[4];
#pragma unroll
  for (int r = 0; r < 8; r++) { mstat[r] = -3.0e38f; lstat[r] = 0.f; }
#pragma unroll
  for (int t = 0; t < 4; t++)
#pragma unroll
    for (int r = 0; r < 8; r++) oacc[t][r] = 0.f;

  const int skk = tid >> 2, scb = (tid & 3) * 16;  // staging: 16 elems/thr

  for (int kb = 0; kb < S; kb += 32) {
    // stage K block [32][64] via async copy (no conversion needed)
    {
      const unsigned short* kp = Kh + (size_t)(kb + skk) * 64 + scb;
      async_b128(&Klds[skk * 72 + scb], kp);
      async_b128(&Klds[skk * 72 + scb + 8], kp + 8);
    }
    // stage V block transposed: Vt[dh][key] (layout change -> manual)
    {
      const unsigned short* vp = Vh + (size_t)(kb + skk) * 64 + scb;
#pragma unroll
      for (int j = 0; j < 16; j++) Vt[(scb + j) * 40 + skk] = vp[j];
    }
    wait_async0();
    __syncthreads();

    // scores = Q @ K^T  (2 n-tiles of 16 keys, 2 k-steps of 32)
    v8f sc[2];
#pragma unroll
    for (int nt = 0; nt < 2; nt++)
#pragma unroll
      for (int r = 0; r < 8; r++) sc[nt][r] = 0.f;
#pragma unroll
    for (int ks = 0; ks < 2; ks++) {
#pragma unroll
      for (int nt = 0; nt < 2; nt++) {
        ABFrag kf;
        const unsigned short* p = &Klds[(nt * 16 + l16) * 72 + ks * 32];
        kf.h[0] = *(const v8bf*)(p + half * 8);
        kf.h[1] = *(const v8bf*)(p + 16 + half * 8);
        sc[nt] = wmma_bf16(qa[ks], kf, sc[nt]);
      }
    }

    // online softmax over these 32 keys
    float rowmax[8];
#pragma unroll
    for (int r = 0; r < 8; r++) {
      float v = fmaxf(sc[0][r], sc[1][r]);
#pragma unroll
      for (int m = 8; m >= 1; m >>= 1) v = fmaxf(v, __shfl_xor(v, m, 32));
      rowmax[r] = v;
    }
#pragma unroll
    for (int r = 0; r < 8; r++) {
      const float mnew = fmaxf(mstat[r], rowmax[r]);
      const float scale = __expf(mstat[r] - mnew);
      float p0 = __expf(sc[0][r] - mnew);
      float p1 = __expf(sc[1][r] - mnew);
      sc[0][r] = p0; sc[1][r] = p1;
      float s = p0 + p1;
#pragma unroll
      for (int m = 8; m >= 1; m >>= 1) s += __shfl_xor(s, m, 32);
      lstat[r] = lstat[r] * scale + s;
      mstat[r] = mnew;
#pragma unroll
      for (int t = 0; t < 4; t++) oacc[t][r] *= scale;
    }

    // P: C-layout regs -> LDS (bf16) -> A-layout fragment (same-wave, in-order)
#pragma unroll
    for (int nt = 0; nt < 2; nt++)
#pragma unroll
      for (int r = 0; r < 8; r++)
        Plds[(wave * 16 + r + 8 * half) * 40 + nt * 16 + l16] = f2bf(sc[nt][r]);

    ABFrag pf;
    {
      const unsigned short* pp = &Plds[(wave * 16 + l16) * 40];
      pf.h[0] = *(const v8bf*)(pp + half * 8);
      pf.h[1] = *(const v8bf*)(pp + 16 + half * 8);
    }
    // O += P @ V  (4 dh-tiles of 16)
#pragma unroll
    for (int t = 0; t < 4; t++) {
      ABFrag vf;
      const unsigned short* p = &Vt[(t * 16 + l16) * 40];
      vf.h[0] = *(const v8bf*)(p + half * 8);
      vf.h[1] = *(const v8bf*)(p + 16 + half * 8);
      oacc[t] = wmma_bf16(pf, vf, oacc[t]);
    }
    __syncthreads();  // before restaging K/V next iteration
  }

  // write O[b][s][h*64+dh] (f32)
#pragma unroll
  for (int r = 0; r < 8; r++) {
    const float inv = 1.0f / lstat[r];
    const int row = qblk * 64 + wave * 16 + r + 8 * half;
#pragma unroll
    for (int t = 0; t < 4; t++) {
      const int dh = t * 16 + l16;
      O[((size_t)bb * S + row) * 1024 + hh * 64 + dh] = oacc[t][r] * inv;
    }
  }
}

// ---------------------------------------------------------------------------
// Row LayerNorm: out = LN(a (+ b)) * g + beta, D=1024, one WG per row.
// ---------------------------------------------------------------------------
__launch_bounds__(256)
__global__ void ln_kernel(const float* __restrict__ a,
                          const float* __restrict__ b,
                          const float* __restrict__ g,
                          const float* __restrict__ beta,
                          float* __restrict__ out, int D) {
  __shared__ float red1[8], red2[8];
  const size_t row = blockIdx.x;
  const int tid = threadIdx.x, wave = tid >> 5, lane = tid & 31;
  float v[4], s = 0.f, s2 = 0.f;
#pragma unroll
  for (int i = 0; i < 4; i++) {
    const int c = tid + i * 256;
    float x = a[row * D + c];
    if (b) x += b[row * D + c];
    v[i] = x; s += x; s2 += x * x;
  }
#pragma unroll
  for (int m = 16; m >= 1; m >>= 1) {
    s += __shfl_xor(s, m, 32);
    s2 += __shfl_xor(s2, m, 32);
  }
  if (lane == 0) { red1[wave] = s; red2[wave] = s2; }
  __syncthreads();
  if (tid == 0) {
    float S = 0.f, S2 = 0.f;
    for (int i = 0; i < 8; i++) { S += red1[i]; S2 += red2[i]; }
    red1[0] = S; red2[0] = S2;
  }
  __syncthreads();
  const float mu = red1[0] / D;
  const float var = red2[0] / D - mu * mu;
  const float rinv = rsqrtf(var + 1e-5f);
#pragma unroll
  for (int i = 0; i < 4; i++) {
    const int c = tid + i * 256;
    out[row * D + c] = (v[i] - mu) * rinv * g[c] + beta[c];
  }
}

// ---------------------------------------------------------------------------
// Host launcher
// ---------------------------------------------------------------------------
extern "C" void kernel_launch(void* const* d_in, const int* in_sizes, int n_in,
                              void* d_out, int out_size, void* d_ws, size_t ws_size,
                              hipStream_t stream) {
  (void)in_sizes; (void)n_in; (void)out_size; (void)ws_size;
  const float* x     = (const float*)d_in[0];
  const float* Wqkv  = (const float*)d_in[1];
  const float* bqkv  = (const float*)d_in[2];
  const float* W1    = (const float*)d_in[3];
  const float* b1    = (const float*)d_in[4];
  const float* W2    = (const float*)d_in[5];
  const float* b2    = (const float*)d_in[6];
  const float* g1    = (const float*)d_in[7];
  const float* beta1 = (const float*)d_in[8];
  const float* g2    = (const float*)d_in[9];
  const float* beta2 = (const float*)d_in[10];

  const int B = 4, S = 4096, D = 1024, H = 16;
  const int M = B * S;  // 16384

  char* ws = (char*)d_ws;
  const size_t MB = 1024 * 1024;
  unsigned short* Qb   = (unsigned short*)(ws);                 // 32 MB bf16
  unsigned short* Kb   = (unsigned short*)(ws + 32 * MB);       // 32 MB
  unsigned short* Vb   = (unsigned short*)(ws + 64 * MB);       // 32 MB
  float*          Obuf = (float*)(ws + 96 * MB);                // 64 MB f32
  float*          x1   = (float*)(ws + 160 * MB);               // 64 MB f32
  unsigned short* hbuf = (unsigned short*)(ws + 224 * MB);      // 128 MB bf16

  // 1) QKV projection -> Q (pre-scaled), K, V in [B,H,S,Dh] bf16
  gemm_wmma<EPI_QKV, false><<<dim3(M / 128, (3 * D) / 128), 256, 0, stream>>>(
      x, Wqkv, bqkv, nullptr, Qb, Kb, Vb, M, 3 * D, D);

  // 2) Flash attention -> Obuf [B,S,D] f32
  flash_attn<<<B * H * (S / 64), 128, 0, stream>>>(Qb, Kb, Vb, Obuf, H, S);

  // 3) x1 = LN(x + attn_out)
  ln_kernel<<<M, 256, 0, stream>>>(x, Obuf, g1, beta1, x1, D);

  // 4) h = gelu(x1 @ W1 + b1) -> bf16
  gemm_wmma<EPI_GELU, false><<<dim3(M / 128, (4 * D) / 128), 256, 0, stream>>>(
      x1, W1, b1, nullptr, hbuf, nullptr, nullptr, M, 4 * D, D);

  // 5) ffn = h @ W2 + b2 + x1 -> Obuf (f32) [A tile via Tensor Data Mover]
  gemm_wmma<EPI_RES, true><<<dim3(M / 128, D / 128), 256, 0, stream>>>(
      hbuf, W2, b2, x1, Obuf, nullptr, nullptr, M, D, 4 * D);

  // 6) out = LN(ffn)
  ln_kernel<<<M, 256, 0, stream>>>(Obuf, nullptr, g2, beta2, (float*)d_out, D);
}
